// TransformerBlock_85736137162886
// MI455X (gfx1250) — matrix operand order repaired and checked by reading, not compiled, not run on hardware
//
#include <hip/hip_runtime.h>

// ---------------------------------------------------------------------------
// CDNA5 (gfx1250, wave32) transformer block.
// All GEMMs + attention einsums via V_WMMA_F32_16X16X32_BF16 (f32 accumulate).
// ---------------------------------------------------------------------------

typedef __attribute__((ext_vector_type(16))) __bf16 v16bf;
typedef __attribute__((ext_vector_type(8)))  float  v8f;

union Frag {
    uint4 q[2];
    v16bf v;
    unsigned short s[16];
};

__device__ __forceinline__ unsigned short f2bf(float f) {
    unsigned int u = __float_as_uint(f);
    u += 0x7FFFu + ((u >> 16) & 1u);   // round-to-nearest-even
    return (unsigned short)(u >> 16);
}

// ---------------------------------------------------------------------------
// fp32 -> bf16 cast (weights)
// ---------------------------------------------------------------------------
__global__ __launch_bounds__(256) void cast_bf16_kernel(
    const float* __restrict__ src, unsigned short* __restrict__ dst, size_t n)
{
    size_t i = (size_t)blockIdx.x * 256 + threadIdx.x;
    size_t stride = (size_t)gridDim.x * 256;
    for (; i < n; i += stride) dst[i] = f2bf(src[i]);
}

// ---------------------------------------------------------------------------
// RMSNorm: one block per row, fp32 in -> bf16 out (scale fused)
// ---------------------------------------------------------------------------
__global__ __launch_bounds__(256) void rmsnorm_bf16_kernel(
    const float* __restrict__ x, const float* __restrict__ scale,
    unsigned short* __restrict__ out, int ncols)
{
    __shared__ float red[8];
    const int row = blockIdx.x;
    const float* xr = x + (size_t)row * ncols;
    float ss = 0.f;
    for (int c = threadIdx.x; c < ncols; c += 256) { float v = xr[c]; ss += v * v; }
    #pragma unroll
    for (int m = 16; m >= 1; m >>= 1) ss += __shfl_xor(ss, m, 32);
    if ((threadIdx.x & 31) == 0) red[threadIdx.x >> 5] = ss;
    __syncthreads();
    float tot = 0.f;
    #pragma unroll
    for (int i = 0; i < 8; i++) tot += red[i];
    const float r = rsqrtf(tot / (float)ncols + 1e-6f);
    unsigned short* orow = out + (size_t)row * ncols;
    for (int c = threadIdx.x; c < ncols; c += 256) orow[c] = f2bf(xr[c] * r * scale[c]);
}

// ---------------------------------------------------------------------------
// bf16 GEMM: C[M,N] (f32) = A[M,K] * B[K,N] (+ resid).  Row-major bf16 inputs.
// Block tile 128x128, BK=32, 8 waves (4 along M, 2 along N), wave tile 32x64.
// B is transposed into LDS so B-fragments are contiguous ds_read_b128 pairs.
// ---------------------------------------------------------------------------
__global__ __launch_bounds__(256) void gemm_bf16_kernel(
    const unsigned short* __restrict__ A, const unsigned short* __restrict__ B,
    const float* __restrict__ resid, float* __restrict__ C,
    int M, int N, int K)
{
    constexpr int BM = 128, BN = 128, BK = 32, PA = 40, PB = 40;
    __shared__ unsigned short sA[BM * PA];
    __shared__ unsigned short sBt[BN * PB];

    const int tid = threadIdx.x, lane = tid & 31, wv = tid >> 5;
    const int hi = lane >> 4, lm = lane & 15;
    const int m0 = blockIdx.x * BM, n0 = blockIdx.y * BN;
    const int wm = (wv & 3) * 32, wn = (wv >> 2) * 64;

    v8f acc[2][4];
    #pragma unroll
    for (int i = 0; i < 2; i++)
        #pragma unroll
        for (int j = 0; j < 4; j++)
            #pragma unroll
            for (int v = 0; v < 8; v++) acc[i][j][v] = 0.f;

    const int ar = tid >> 1, acol = (tid & 1) * 16;   // A tile loader
    const int bkr = tid >> 3, bn = (tid & 7) * 16;    // B tile loader

    for (int kb = 0; kb < K; kb += BK) {
        __syncthreads();
        // stage A tile (row-major)
        const unsigned short* ga = A + (size_t)(m0 + ar) * K + kb + acol;
        *(uint4*)&sA[ar * PA + acol]     = *(const uint4*)ga;
        *(uint4*)&sA[ar * PA + acol + 8] = *(const uint4*)(ga + 8);
        // stage B tile transposed: sBt[n][k]
        const unsigned short* gb = B + (size_t)(kb + bkr) * N + n0 + bn;
        Frag bu;
        bu.q[0] = *(const uint4*)gb;
        bu.q[1] = *(const uint4*)(gb + 8);
        #pragma unroll
        for (int i = 0; i < 16; i++) sBt[(bn + i) * PB + bkr] = bu.s[i];
        if (kb + BK < K) {
            // locality 3 -> near (WGP) scope: pull next K-tile into all cache levels
            __builtin_prefetch(ga + BK, 0, 3);                                  // global_prefetch_b8
            __builtin_prefetch(B + (size_t)(kb + BK + bkr) * N + n0 + bn, 0, 3);
        }
        __syncthreads();

        // A fragments: m = lm, k = e + 16*hi (contiguous 32B, same k order as B)
        Frag af[2], bf[4];
        #pragma unroll
        for (int i = 0; i < 2; i++) {
            const unsigned short* p = &sA[(wm + i * 16 + lm) * PA + hi * 16];
            af[i].q[0] = *(const uint4*)p;
            af[i].q[1] = *(const uint4*)(p + 8);
        }
        // B fragments: n = lm, k = e + 16*hi (contiguous 32B)
        #pragma unroll
        for (int j = 0; j < 4; j++) {
            const unsigned short* p = &sBt[(wn + j * 16 + lm) * PB + hi * 16];
            bf[j].q[0] = *(const uint4*)p;
            bf[j].q[1] = *(const uint4*)(p + 8);
        }
        #pragma unroll
        for (int i = 0; i < 2; i++)
            #pragma unroll
            for (int j = 0; j < 4; j++)
                acc[i][j] = __builtin_amdgcn_wmma_f32_16x16x32_bf16(
                    false, af[i].v, false, bf[j].v, (short)0, acc[i][j], false, false);
    }

    // epilogue: C layout row = v + 8*hi, col = lm
    #pragma unroll
    for (int i = 0; i < 2; i++)
        #pragma unroll
        for (int j = 0; j < 4; j++)
            #pragma unroll
            for (int v = 0; v < 8; v++) {
                const int m = m0 + wm + i * 16 + v + 8 * hi;
                const int n = n0 + wn + j * 16 + lm;
                const size_t idx = (size_t)m * N + n;
                float val = acc[i][j][v];
                if (resid) val += resid[idx];
                C[idx] = val;
            }
}

// ---------------------------------------------------------------------------
// Fused SwiGLU GEMM: Gm[M,N] (bf16) = silu(A*B1) * (A*B2).
// Block tile 128x64, 8 waves (4x2), wave tile 32x32, two accumulator banks.
// ---------------------------------------------------------------------------
__global__ __launch_bounds__(256) void gemm_swiglu_kernel(
    const unsigned short* __restrict__ A, const unsigned short* __restrict__ B1,
    const unsigned short* __restrict__ B2, unsigned short* __restrict__ Gm,
    int M, int N, int K)
{
    constexpr int BM = 128, BN = 64, BK = 32, PA = 40, PB = 40;
    __shared__ unsigned short sA[BM * PA];
    __shared__ unsigned short sB1[BN * PB];
    __shared__ unsigned short sB2[BN * PB];

    const int tid = threadIdx.x, lane = tid & 31, wv = tid >> 5;
    const int hi = lane >> 4, lm = lane & 15;
    const int m0 = blockIdx.x * BM, n0 = blockIdx.y * BN;
    const int wm = (wv & 3) * 32, wn = (wv >> 2) * 32;

    v8f ac1[2][2], ac2[2][2];
    #pragma unroll
    for (int i = 0; i < 2; i++)
        #pragma unroll
        for (int j = 0; j < 2; j++)
            #pragma unroll
            for (int v = 0; v < 8; v++) { ac1[i][j][v] = 0.f; ac2[i][j][v] = 0.f; }

    const int ar = tid >> 1, acol = (tid & 1) * 16;
    const int bkr = tid >> 3, bn = (tid & 7) * 8;

    for (int kb = 0; kb < K; kb += BK) {
        __syncthreads();
        const unsigned short* ga = A + (size_t)(m0 + ar) * K + kb + acol;
        *(uint4*)&sA[ar * PA + acol]     = *(const uint4*)ga;
        *(uint4*)&sA[ar * PA + acol + 8] = *(const uint4*)(ga + 8);

        union { uint4 q; unsigned short s[8]; } u1, u2;
        u1.q = *(const uint4*)(B1 + (size_t)(kb + bkr) * N + n0 + bn);
        u2.q = *(const uint4*)(B2 + (size_t)(kb + bkr) * N + n0 + bn);
        #pragma unroll
        for (int i = 0; i < 8; i++) {
            sB1[(bn + i) * PB + bkr] = u1.s[i];
            sB2[(bn + i) * PB + bkr] = u2.s[i];
        }
        if (kb + BK < K) {
            __builtin_prefetch(ga + BK, 0, 3);
            __builtin_prefetch(B1 + (size_t)(kb + BK + bkr) * N + n0 + bn, 0, 3);
            __builtin_prefetch(B2 + (size_t)(kb + BK + bkr) * N + n0 + bn, 0, 3);
        }
        __syncthreads();

        Frag af[2], b1f[2], b2f[2];
        #pragma unroll
        for (int i = 0; i < 2; i++) {
            const unsigned short* p = &sA[(wm + i * 16 + lm) * PA + hi * 16];
            af[i].q[0] = *(const uint4*)p;
            af[i].q[1] = *(const uint4*)(p + 8);
        }
        #pragma unroll
        for (int j = 0; j < 2; j++) {
            const unsigned short* p1 = &sB1[(wn + j * 16 + lm) * PB + hi * 16];
            b1f[j].q[0] = *(const uint4*)p1; b1f[j].q[1] = *(const uint4*)(p1 + 8);
            const unsigned short* p2 = &sB2[(wn + j * 16 + lm) * PB + hi * 16];
            b2f[j].q[0] = *(const uint4*)p2; b2f[j].q[1] = *(const uint4*)(p2 + 8);
        }
        #pragma unroll
        for (int i = 0; i < 2; i++)
            #pragma unroll
            for (int j = 0; j < 2; j++) {
                ac1[i][j] = __builtin_amdgcn_wmma_f32_16x16x32_bf16(
                    false, af[i].v, false, b1f[j].v, (short)0, ac1[i][j], false, false);
                ac2[i][j] = __builtin_amdgcn_wmma_f32_16x16x32_bf16(
                    false, af[i].v, false, b2f[j].v, (short)0, ac2[i][j], false, false);
            }
    }

    #pragma unroll
    for (int i = 0; i < 2; i++)
        #pragma unroll
        for (int j = 0; j < 2; j++)
            #pragma unroll
            for (int v = 0; v < 8; v++) {
                const int m = m0 + wm + i * 16 + v + 8 * hi;
                const int n = n0 + wn + j * 16 + lm;
                const float u = ac1[i][j][v];
                const float g = (u / (1.f + __expf(-u))) * ac2[i][j][v];
                Gm[(size_t)m * N + n] = f2bf(g);
            }
}

// ---------------------------------------------------------------------------
// Split qkv (fp32, [tok, 3D]) into bf16 Q[bh,T,hd] (pre-scaled by 1/sqrt(hd)),
// K[bh,T,hd], and transposed Vt[bh,hd,T] (so PV B-fragments are contiguous).
// ---------------------------------------------------------------------------
__global__ __launch_bounds__(256) void split_qkv_kernel(
    const float* __restrict__ qkv, unsigned short* __restrict__ Qo,
    unsigned short* __restrict__ Ko, unsigned short* __restrict__ Vt)
{
    constexpr int T = 2048, HD = 128, H = 16, D = 2048;
    const int token = blockIdx.x;
    const int b = token / T, t = token % T;
    const float* row = qkv + (size_t)token * (3 * D);
    const float qscale = 0.08838834764831845f;  // 1/sqrt(128)
    for (int c = threadIdx.x; c < D; c += 256) {
        const int h = c / HD, d = c % HD;
        const size_t bh = (size_t)(b * H + h);
        Qo[(bh * T + t) * HD + d] = f2bf(row[c] * qscale);
        Ko[(bh * T + t) * HD + d] = f2bf(row[D + c]);
        Vt[(bh * HD + d) * T + t] = f2bf(row[2 * D + c]);
    }
}

// ---------------------------------------------------------------------------
// Flash attention: grid (B*H, T/64), 128 threads = 4 waves, 16 query rows/wave.
// S = Q*K^T via WMMA (Q frags from global, K rows are B-fragment contiguous),
// online softmax (16-lane shfl_xor reductions), P -> LDS -> A-fragments,
// O += P*V via WMMA with transposed Vt.  Output y bf16 [tok, D].
// ---------------------------------------------------------------------------
__global__ __launch_bounds__(128) void fattn_kernel(
    const unsigned short* __restrict__ Q, const unsigned short* __restrict__ Kb,
    const unsigned short* __restrict__ Vt, unsigned short* __restrict__ Y)
{
    constexpr int T = 2048, HD = 128, H = 16, D = 2048;
    __shared__ unsigned short sP[4][16 * 64];

    const int lane = threadIdx.x & 31, wv = threadIdx.x >> 5;
    const int hi = lane >> 4, lm = lane & 15;
    const int bh = blockIdx.x, qt = blockIdx.y;

    // Q A-fragments for this wave's 16 rows (held for whole kernel)
    const unsigned short* qb = Q + ((size_t)bh * T + qt * 64 + wv * 16 + lm) * HD;
    Frag qf[4];
    #pragma unroll
    for (int f = 0; f < 4; f++) {
        const unsigned short* p = qb + f * 32 + hi * 16;
        qf[f].q[0] = *(const uint4*)p;
        qf[f].q[1] = *(const uint4*)(p + 8);
    }

    float mi[8], li[8];
    v8f o[8];
    #pragma unroll
    for (int v = 0; v < 8; v++) { mi[v] = -3.0e38f; li[v] = 0.f; }
    #pragma unroll
    for (int no = 0; no < 8; no++)
        #pragma unroll
        for (int v = 0; v < 8; v++) o[no][v] = 0.f;

    for (int jt = 0; jt < T / 64; jt++) {
        // ---- S = Q * K^T (16x64) ----
        v8f s[4];
        #pragma unroll
        for (int ns = 0; ns < 4; ns++)
            #pragma unroll
            for (int v = 0; v < 8; v++) s[ns][v] = 0.f;
        #pragma unroll
        for (int ns = 0; ns < 4; ns++) {
            const unsigned short* kp =
                Kb + ((size_t)bh * T + jt * 64 + ns * 16 + lm) * HD + hi * 16;
            #pragma unroll
            for (int f = 0; f < 4; f++) {
                Frag bf;
                const unsigned short* p = kp + f * 32;
                bf.q[0] = *(const uint4*)p;
                bf.q[1] = *(const uint4*)(p + 8);
                s[ns] = __builtin_amdgcn_wmma_f32_16x16x32_bf16(
                    false, qf[f].v, false, bf.v, (short)0, s[ns], false, false);
            }
        }
        // ---- online softmax (row = v + 8*hi; reduce over 16-lane halves) ----
        #pragma unroll
        for (int v = 0; v < 8; v++) {
            float mx = fmaxf(fmaxf(s[0][v], s[1][v]), fmaxf(s[2][v], s[3][v]));
            #pragma unroll
            for (int msk = 1; msk < 16; msk <<= 1) mx = fmaxf(mx, __shfl_xor(mx, msk, 32));
            const float mn = fmaxf(mi[v], mx);
            const float sc = __expf(mi[v] - mn);
            mi[v] = mn;
            float sum = 0.f;
            #pragma unroll
            for (int ns = 0; ns < 4; ns++) {
                const float e = __expf(s[ns][v] - mn);
                s[ns][v] = e;
                sum += e;
            }
            #pragma unroll
            for (int msk = 1; msk < 16; msk <<= 1) sum += __shfl_xor(sum, msk, 32);
            li[v] = li[v] * sc + sum;
            #pragma unroll
            for (int no = 0; no < 8; no++) o[no][v] *= sc;
        }
        // ---- P (C-layout) -> LDS -> A-fragments ----
        #pragma unroll
        for (int ns = 0; ns < 4; ns++)
            #pragma unroll
            for (int v = 0; v < 8; v++)
                sP[wv][(v + 8 * hi) * 64 + ns * 16 + lm] = f2bf(s[ns][v]);
        __syncthreads();
        Frag pf[2];
        #pragma unroll
        for (int f = 0; f < 2; f++) {
            const unsigned short* p = &sP[wv][lm * 64 + f * 32 + hi * 16];
            pf[f].q[0] = *(const uint4*)p;
            pf[f].q[1] = *(const uint4*)(p + 8);
        }
        // ---- O += P * V ----
        #pragma unroll
        for (int no = 0; no < 8; no++) {
            const unsigned short* vp =
                Vt + ((size_t)bh * HD + no * 16 + lm) * T + jt * 64 + hi * 16;
            #pragma unroll
            for (int f = 0; f < 2; f++) {
                Frag bf;
                const unsigned short* p = vp + f * 32;
                bf.q[0] = *(const uint4*)p;
                bf.q[1] = *(const uint4*)(p + 8);
                o[no] = __builtin_amdgcn_wmma_f32_16x16x32_bf16(
                    false, pf[f].v, false, bf.v, (short)0, o[no], false, false);
            }
        }
        __syncthreads();
    }

    // ---- write y[token, h*hd + d] = O / l ----
    const int b = bh / H, h = bh % H;
    #pragma unroll
    for (int no = 0; no < 8; no++)
        #pragma unroll
        for (int v = 0; v < 8; v++) {
            const int trow = qt * 64 + wv * 16 + v + 8 * hi;
            const size_t token = (size_t)b * T + trow;
            const int col = h * HD + no * 16 + lm;
            Y[token * D + col] = f2bf(o[no][v] / li[v]);
        }
}

// ---------------------------------------------------------------------------
// Launch
// ---------------------------------------------------------------------------
extern "C" void kernel_launch(void* const* d_in, const int* in_sizes, int n_in,
                              void* d_out, int out_size, void* d_ws, size_t ws_size,
                              hipStream_t stream)
{
    constexpr int D = 2048, H = 16, HID = 8192, B = 2, T = 2048, TOK = B * T;
    (void)in_sizes; (void)n_in; (void)out_size; (void)ws_size;

    const float* x      = (const float*)d_in[0];
    const float* scale1 = (const float*)d_in[1];
    const float* scale2 = (const float*)d_in[2];
    const float* w_qkv  = (const float*)d_in[3];
    const float* w_o    = (const float*)d_in[4];
    const float* w1     = (const float*)d_in[5];
    const float* w2     = (const float*)d_in[6];
    const float* w3     = (const float*)d_in[7];

    char* p = (char*)d_ws;
    auto alloc = [&](size_t bytes) {
        void* r = (void*)p;
        p += (bytes + 255) & ~(size_t)255;
        return r;
    };
    unsigned short* wqkv_bf = (unsigned short*)alloc((size_t)D * 3 * D * 2);
    unsigned short* wo_bf   = (unsigned short*)alloc((size_t)D * D * 2);
    unsigned short* w1_bf   = (unsigned short*)alloc((size_t)D * HID * 2);
    unsigned short* w2_bf   = (unsigned short*)alloc((size_t)D * HID * 2);
    unsigned short* w3_bf   = (unsigned short*)alloc((size_t)HID * D * 2);
    unsigned short* h_bf    = (unsigned short*)alloc((size_t)TOK * D * 2);
    float*          qkv_f   = (float*)alloc((size_t)TOK * 3 * D * 4);
    unsigned short* q_bf    = (unsigned short*)alloc((size_t)TOK * D * 2);
    unsigned short* k_bf    = (unsigned short*)alloc((size_t)TOK * D * 2);
    unsigned short* vt_bf   = (unsigned short*)alloc((size_t)TOK * D * 2);
    unsigned short* y_bf    = (unsigned short*)alloc((size_t)TOK * D * 2);
    float*          x1_f    = (float*)alloc((size_t)TOK * D * 4);
    unsigned short* h2_bf   = (unsigned short*)alloc((size_t)TOK * D * 2);
    unsigned short* gm_bf   = (unsigned short*)alloc((size_t)TOK * HID * 2);

    // weight casts (recomputed each call: deterministic, no state)
    cast_bf16_kernel<<<2048, 256, 0, stream>>>(w_qkv, wqkv_bf, (size_t)D * 3 * D);
    cast_bf16_kernel<<<2048, 256, 0, stream>>>(w_o,   wo_bf,   (size_t)D * D);
    cast_bf16_kernel<<<2048, 256, 0, stream>>>(w1,    w1_bf,   (size_t)D * HID);
    cast_bf16_kernel<<<2048, 256, 0, stream>>>(w2,    w2_bf,   (size_t)D * HID);
    cast_bf16_kernel<<<2048, 256, 0, stream>>>(w3,    w3_bf,   (size_t)HID * D);

    // attention sublayer
    rmsnorm_bf16_kernel<<<TOK, 256, 0, stream>>>(x, scale1, h_bf, D);
    gemm_bf16_kernel<<<dim3(TOK / 128, 3 * D / 128), 256, 0, stream>>>(
        h_bf, wqkv_bf, nullptr, qkv_f, TOK, 3 * D, D);
    split_qkv_kernel<<<TOK, 256, 0, stream>>>(qkv_f, q_bf, k_bf, vt_bf);
    fattn_kernel<<<dim3(B * H, T / 64), 128, 0, stream>>>(q_bf, k_bf, vt_bf, y_bf);
    gemm_bf16_kernel<<<dim3(TOK / 128, D / 128), 256, 0, stream>>>(
        y_bf, wo_bf, x, x1_f, TOK, D, D);   // resid = x

    // SwiGLU MLP sublayer
    rmsnorm_bf16_kernel<<<TOK, 256, 0, stream>>>(x1_f, scale2, h2_bf, D);
    gemm_swiglu_kernel<<<dim3(TOK / 128, HID / 64), 256, 0, stream>>>(
        h2_bf, w1_bf, w2_bf, gm_bf, TOK, HID, D);
    gemm_bf16_kernel<<<dim3(TOK / 128, D / 128), 256, 0, stream>>>(
        gm_bf, w3_bf, x1_f, (float*)d_out, TOK, D, HID);  // resid = x1
}
